// StockPriceHybridModel_82867099009218
// MI455X (gfx1250) — matrix operand order repaired and checked by reading, not compiled, not run on hardware
//
#include <hip/hip_runtime.h>

typedef __attribute__((ext_vector_type(16))) __bf16 v16bf;
typedef __attribute__((ext_vector_type(8)))  __bf16 v8bf;
typedef __attribute__((ext_vector_type(8)))  float  v8f;

constexpr int T_ = 64;   // timesteps
constexpr int F_ = 32;   // input features
constexpr int H_ = 64;   // hidden
constexpr int HS_ = H_ + 8;  // padded LDS row stride (bf16): 144B = 36 banks -> conflict-free
constexpr int G1_ = 64;  // gcn1 out
constexpr int G2_ = 32;  // gcn2 out

#define DEV static __device__ __forceinline__

DEV float bf2f(__bf16 b) {
  unsigned short s = __builtin_bit_cast(unsigned short, b);
  unsigned u = ((unsigned)s) << 16;
  return __builtin_bit_cast(float, u);
}
DEV __bf16 f2bf(float f) {
  unsigned u = __builtin_bit_cast(unsigned, f);
  u += 0x7FFFu + ((u >> 16) & 1u);            // round-to-nearest-even
  unsigned short s = (unsigned short)(u >> 16);
  return __builtin_bit_cast(__bf16, s);
}
DEV float sigm(float v)      { return 1.0f / (1.0f + __expf(-v)); }
DEV float tanh_fast(float v) { return 1.0f - 2.0f / (1.0f + __expf(2.0f * v)); }

// Opaque zero offset: makes each weight-fragment load's address a fresh value so
// the compiler cannot hoist/CSE the (loop-invariant) loads out of the timestep
// loop (hoisting all 112 B-fragments = ~896 VGPRs -> scratch spills). Laundering
// an *integer* (not the pointer) preserves the global address-space provenance
// of W, so the loads lower to global_load_b128 (LOADcnt only) instead of
// flat_load_b128 (which also ties up DScnt / the LDS aperture path).
DEV unsigned launder_zero() {
  unsigned z = 0;
  asm volatile("" : "+v"(z));
  return z;
}

// ---- WMMA fragment helpers (v_wmma_f32_16x16x32_bf16, wave32) -------------
// A (16x32, MxK): lanes 0-15 hold row m=lane, K in {kb..kb+7} u {16+kb..16+kb+7},
// kb = 0 for lanes 0-15, 8 for lanes 16-31 (ISA 7.12.2 16-bit A layout).
DEV v16bf a_from_global_f32(const float* __restrict__ rp, int lane) {
  const int kb = (lane & 16) ? 8 : 0;
  v16bf a;
#pragma unroll
  for (int e = 0; e < 8; ++e) {
    a[e]     = f2bf(rp[kb + e]);
    a[e + 8] = f2bf(rp[kb + 16 + e]);
  }
  return a;
}
// A slice (K-chunk kc in {0,32}) from a padded LDS bf16 row (stride HS_).
DEV v16bf a_from_lds(const __bf16* rp, int lane, int kc) {
  const int kb = kc + ((lane & 16) ? 8 : 0);
  v8bf lo = *(const v8bf*)(rp + kb);
  v8bf hi = *(const v8bf*)(rp + kb + 16);
  v16bf a;
#pragma unroll
  for (int e = 0; e < 8; ++e) { a[e] = lo[e]; a[e + 8] = hi[e]; }
  return a;
}
// B (32x16, KxN): lane holds column n = lane&15; K in {kb..kb+7} u {16+kb..16+kb+7},
// kb = kc + (0 for lanes 0-15, 8 for lanes 16-31). Weights stored [256][K] row-major bf16 in global
// (hot in WGP$/L2: all four matrices total 112KB); two 16B pieces per lane.
DEV v16bf b_from_w(const __bf16* __restrict__ W, int K, int nt, int kc, int lane) {
  const int n = 16 * nt + (lane & 15);
  const int k = kc + ((lane & 16) ? 8 : 0);
  const __bf16* p = W + (size_t)n * K + k + launder_zero();
  v8bf lo = *(const v8bf*)(p);
  v8bf hi = *(const v8bf*)(p + 16);
  v16bf b;
#pragma unroll
  for (int e = 0; e < 8; ++e) { b[e] = lo[e]; b[e + 8] = hi[e]; }
  return b;
}
DEV v8f wmma_bf16(v16bf a, v16bf b, v8f c) {
  return __builtin_amdgcn_wmma_f32_16x16x32_bf16(false, a, false, b, (short)0, c,
                                                 false, false);
}

// ---------------------------------------------------------------------------
// Fused 2-layer LSTM. Block = 128 threads = 4 waves; wave owns 16 batch rows.
// h-state (bf16) lives in wave-private padded LDS rows; c-state fp32 in regs.
// Writes final-timestep layer-1 h (fp32) to h2out.
// ---------------------------------------------------------------------------
__global__ void __launch_bounds__(128) lstm2_fused(
    const float* __restrict__ x,
    const __bf16* __restrict__ Wih0, const __bf16* __restrict__ Whh0,
    const __bf16* __restrict__ Wih1, const __bf16* __restrict__ Whh1,
    const float* __restrict__ b0, const float* __restrict__ b1,
    float* __restrict__ h2out, int N) {
  __shared__ __bf16 h0_lds[64 * HS_];
  __shared__ __bf16 h1_lds[64 * HS_];

  const int lane = threadIdx.x & 31;
  const int wave = threadIdx.x >> 5;
  const int m    = lane & 15;
  const int half = lane >> 4;
  const int col  = m;                          // D-fragment column for this lane
  const int rowBase = blockIdx.x * 64 + wave * 16;

  // zero this wave's h rows (wave-private -> no block barrier needed anywhere)
  for (int i = lane; i < 16 * HS_; i += 32) {
    h0_lds[wave * 16 * HS_ + i] = f2bf(0.0f);
    h1_lds[wave * 16 * HS_ + i] = f2bf(0.0f);
  }

  int rowA = rowBase + m;
  if (rowA > N - 1) rowA = N - 1;              // clamp: keep EXEC all-ones for WMMA
  const float* xrow = x + (size_t)rowA * T_ * F_;

  const __bf16* h0row = h0_lds + (wave * 16 + m) * HS_;
  const __bf16* h1row = h1_lds + (wave * 16 + m) * HS_;
  __bf16* h0w = h0_lds + wave * 16 * HS_;
  __bf16* h1w = h1_lds + wave * 16 * HS_;

  v8f c0[4], c1[4];
#pragma unroll
  for (int j = 0; j < 4; ++j) { c0[j] = {}; c1[j] = {}; }

#pragma unroll 1
  for (int t = 0; t < T_; ++t) {
    // ----- layer 0: gates = x_t @ Wih0^T + h0 @ Whh0^T + b -----
    v16bf ax   = a_from_global_f32(xrow + t * F_, lane);
    v16bf ah0a = a_from_lds(h0row, lane, 0);
    v16bf ah0b = a_from_lds(h0row, lane, 32);
#pragma unroll
    for (int jt = 0; jt < 4; ++jt) {
      v8f g[4];
#pragma unroll
      for (int q = 0; q < 4; ++q) {            // q: i,f,g,o gate group
        const int nt = jt + 4 * q;             // gate column tile
        v8f acc = {};
        acc = wmma_bf16(ax,   b_from_w(Wih0, F_, nt, 0,  lane), acc);
        acc = wmma_bf16(ah0a, b_from_w(Whh0, H_, nt, 0,  lane), acc);
        acc = wmma_bf16(ah0b, b_from_w(Whh0, H_, nt, 32, lane), acc);
        const float bv = b0[16 * nt + col];
#pragma unroll
        for (int r = 0; r < 8; ++r) acc[r] += bv;
        g[q] = acc;
      }
#pragma unroll
      for (int r = 0; r < 8; ++r) {
        float iv = sigm(g[0][r]);
        float fv = sigm(g[1][r]);
        float gv = tanh_fast(g[2][r]);
        float ov = sigm(g[3][r]);
        float cv = fv * c0[jt][r] + iv * gv;
        c0[jt][r] = cv;
        float hv = ov * tanh_fast(cv);
        h0w[(r + 8 * half) * HS_ + 16 * jt + col] = f2bf(hv);
      }
    }
    // ----- layer 1: gates = h0_new @ Wih1^T + h1 @ Whh1^T + b -----
    v16bf an0  = a_from_lds(h0row, lane, 0);
    v16bf an1  = a_from_lds(h0row, lane, 32);
    v16bf ah1a = a_from_lds(h1row, lane, 0);
    v16bf ah1b = a_from_lds(h1row, lane, 32);
#pragma unroll
    for (int jt = 0; jt < 4; ++jt) {
      v8f g[4];
#pragma unroll
      for (int q = 0; q < 4; ++q) {
        const int nt = jt + 4 * q;
        v8f acc = {};
        acc = wmma_bf16(an0,  b_from_w(Wih1, H_, nt, 0,  lane), acc);
        acc = wmma_bf16(an1,  b_from_w(Wih1, H_, nt, 32, lane), acc);
        acc = wmma_bf16(ah1a, b_from_w(Whh1, H_, nt, 0,  lane), acc);
        acc = wmma_bf16(ah1b, b_from_w(Whh1, H_, nt, 32, lane), acc);
        const float bv = b1[16 * nt + col];
#pragma unroll
        for (int r = 0; r < 8; ++r) acc[r] += bv;
        g[q] = acc;
      }
#pragma unroll
      for (int r = 0; r < 8; ++r) {
        float iv = sigm(g[0][r]);
        float fv = sigm(g[1][r]);
        float gv = tanh_fast(g[2][r]);
        float ov = sigm(g[3][r]);
        float cv = fv * c1[jt][r] + iv * gv;
        c1[jt][r] = cv;
        float hv = ov * tanh_fast(cv);
        h1w[(r + 8 * half) * HS_ + 16 * jt + col] = f2bf(hv);
        if (t == T_ - 1) {
          int row = rowBase + r + 8 * half;
          if (row < N) h2out[(size_t)row * H_ + 16 * jt + col] = hv;
        }
      }
    }
  }
}

// ---------------------------------------------------------------------------
// Setup: convert LSTM weights to bf16 (layout unchanged: [4H][K]) and fold
// the two bias vectors per layer.
// ---------------------------------------------------------------------------
__global__ void prep_weights(const float* __restrict__ wih0, const float* __restrict__ whh0,
                             const float* __restrict__ wih1, const float* __restrict__ whh1,
                             const float* __restrict__ bih0, const float* __restrict__ bhh0,
                             const float* __restrict__ bih1, const float* __restrict__ bhh1,
                             __bf16* o_ih0, __bf16* o_hh0, __bf16* o_ih1, __bf16* o_hh1,
                             float* b0, float* b1) {
  int i = blockIdx.x * blockDim.x + threadIdx.x;
  if (i < 4 * H_ * F_) o_ih0[i] = f2bf(wih0[i]);
  if (i < 4 * H_ * H_) {
    o_hh0[i] = f2bf(whh0[i]);
    o_ih1[i] = f2bf(wih1[i]);
    o_hh1[i] = f2bf(whh1[i]);
  }
  if (i < 4 * H_) { b0[i] = bih0[i] + bhh0[i]; b1[i] = bih1[i] + bhh1[i]; }
}

__global__ void zero_f32(float* p, long n) {
  long i = (long)blockIdx.x * blockDim.x + threadIdx.x;
  if (i < n) p[i] = 0.0f;
}
__global__ void deg_init(float* deg, int n) {
  int i = blockIdx.x * blockDim.x + threadIdx.x;
  if (i < n) deg[i] = 1.0f;                    // self-loop
}
__global__ void deg_accum(const int* __restrict__ dst, float* deg, long E) {
  long e = (long)blockIdx.x * blockDim.x + threadIdx.x;
  if (e < E) atomicAdd(&deg[dst[e]], 1.0f);
}
__global__ void deg_rsqrt(float* deg, int n) {
  int i = blockIdx.x * blockDim.x + threadIdx.x;
  if (i < n) deg[i] = rsqrtf(deg[i]);          // deg >= 1 always
}

// out[n][j] = sum_k in[n][k] * W[k][j]   (W row-major [Gi][Go])
__global__ void small_matmul(const float* __restrict__ in, const float* __restrict__ W,
                             float* __restrict__ out, int n, int Gi, int Go) {
  long tid = (long)blockIdx.x * blockDim.x + threadIdx.x;
  if (tid >= (long)n * Go) return;
  int row = (int)(tid / Go), j = (int)(tid % Go);
  const float* ir = in + (size_t)row * Gi;
  float s = 0.f;
  for (int k = 0; k < Gi; ++k) s = fmaf(ir[k], W[k * Go + j], s);
  out[tid] = s;
}

// one thread = one (edge, 4-feature chunk): out[dst] += xw[src]*dinv[s]*dinv[d]
__global__ void gcn_scatter(const int* __restrict__ src, const int* __restrict__ dst,
                            const float* __restrict__ xw, const float* __restrict__ dinv,
                            float* __restrict__ out, long E, int G) {
  long tid = (long)blockIdx.x * blockDim.x + threadIdx.x;
  int per = G >> 2;
  long e = tid / per;
  if (e >= E) return;
  int fc = (int)(tid % per) * 4;
  int s = src[e], d = dst[e];
  float nrm = dinv[s] * dinv[d];
  const float4 v = *(const float4*)(xw + (size_t)s * G + fc);
  float* o = out + (size_t)d * G + fc;
  atomicAdd(o + 0, v.x * nrm);
  atomicAdd(o + 1, v.y * nrm);
  atomicAdd(o + 2, v.z * nrm);
  atomicAdd(o + 3, v.w * nrm);
}

// acc += self-loop term + bias, then ReLU (in place)
__global__ void gcn_finalize(float* __restrict__ acc, const float* __restrict__ xw,
                             const float* __restrict__ dinv, const float* __restrict__ bias,
                             int n, int G) {
  long tid = (long)blockIdx.x * blockDim.x + threadIdx.x;
  if (tid >= (long)n * G) return;
  int row = (int)(tid / G), j = (int)(tid % G);
  float di = dinv[row];
  float v = acc[tid] + xw[tid] * di * di + bias[j];
  acc[tid] = v > 0.f ? v : 0.f;
}

__global__ void fc_gather(const int* __restrict__ ids, const float* __restrict__ g2,
                          const float* __restrict__ fcW, const float* __restrict__ fcb,
                          float* __restrict__ out, int n) {
  int i = blockIdx.x * blockDim.x + threadIdx.x;
  if (i >= n) return;
  const float* r = g2 + (size_t)ids[i] * G2_;
  float s = fcb[0];
#pragma unroll
  for (int j = 0; j < G2_; ++j) s = fmaf(r[j], fcW[j], s);
  out[i] = s;
}

// ---------------------------------------------------------------------------
extern "C" void kernel_launch(void* const* d_in, const int* in_sizes, int n_in,
                              void* d_out, int out_size, void* d_ws, size_t ws_size,
                              hipStream_t stream) {
  const float* x    = (const float*)d_in[0];
  const int*   ids  = (const int*)  d_in[1];
  const int*   edge = (const int*)  d_in[2];
  const float* wih0 = (const float*)d_in[3];
  const float* whh0 = (const float*)d_in[4];
  const float* bih0 = (const float*)d_in[5];
  const float* bhh0 = (const float*)d_in[6];
  const float* wih1 = (const float*)d_in[7];
  const float* whh1 = (const float*)d_in[8];
  const float* bih1 = (const float*)d_in[9];
  const float* bhh1 = (const float*)d_in[10];
  const float* g1W  = (const float*)d_in[11];
  const float* g1b  = (const float*)d_in[12];
  const float* g2W  = (const float*)d_in[13];
  const float* g2b  = (const float*)d_in[14];
  const float* fcW  = (const float*)d_in[15];
  const float* fcb  = (const float*)d_in[16];

  const int  N = in_sizes[1];
  const long E = in_sizes[2] / 2;
  const int* src = edge;
  const int* dst = edge + E;

  char* ws = (char*)d_ws;
  size_t off = 0;
  auto take = [&](size_t bytes) -> char* {
    char* p = ws + off;
    off = (off + bytes + 255) & ~(size_t)255;
    return p;
  };
  __bf16* wb_ih0 = (__bf16*)take((size_t)4 * H_ * F_ * sizeof(__bf16));
  __bf16* wb_hh0 = (__bf16*)take((size_t)4 * H_ * H_ * sizeof(__bf16));
  __bf16* wb_ih1 = (__bf16*)take((size_t)4 * H_ * H_ * sizeof(__bf16));
  __bf16* wb_hh1 = (__bf16*)take((size_t)4 * H_ * H_ * sizeof(__bf16));
  float* b0   = (float*)take((size_t)4 * H_ * sizeof(float));
  float* b1   = (float*)take((size_t)4 * H_ * sizeof(float));
  float* h2   = (float*)take((size_t)N * H_ * sizeof(float));   // lstm out / g2 acc
  float* bufA = (float*)take((size_t)N * H_ * sizeof(float));   // xw scratch
  float* bufB = (float*)take((size_t)N * H_ * sizeof(float));   // g1 acc/out
  float* dinv = (float*)take((size_t)N * sizeof(float));

  // 1. weights -> bf16, fold biases
  prep_weights<<<64, 256, 0, stream>>>(wih0, whh0, wih1, whh1, bih0, bhh0, bih1, bhh1,
                                       wb_ih0, wb_hh0, wb_ih1, wb_hh1, b0, b1);
  // 2. fused 2-layer LSTM (WMMA bf16, fp32 state), h2 = h2[:, -1, :]
  lstm2_fused<<<(N + 63) / 64, 128, 0, stream>>>(x, wb_ih0, wb_hh0, wb_ih1, wb_hh1,
                                                 b0, b1, h2, N);
  // 3. degrees -> dinv
  deg_init <<<(N + 255) / 256, 256, 0, stream>>>(dinv, N);
  deg_accum<<<(int)((E + 255) / 256), 256, 0, stream>>>(dst, dinv, E);
  deg_rsqrt<<<(N + 255) / 256, 256, 0, stream>>>(dinv, N);
  // 4. GCN layer 1: bufB = relu(scatter(h2 @ W1) + self + b1)
  small_matmul<<<(int)(((long)N * G1_ + 255) / 256), 256, 0, stream>>>(h2, g1W, bufA, N, H_, G1_);
  zero_f32    <<<(int)(((long)N * G1_ + 255) / 256), 256, 0, stream>>>(bufB, (long)N * G1_);
  gcn_scatter <<<(int)((E * (G1_ / 4) + 255) / 256), 256, 0, stream>>>(src, dst, bufA, dinv, bufB, E, G1_);
  gcn_finalize<<<(int)(((long)N * G1_ + 255) / 256), 256, 0, stream>>>(bufB, bufA, dinv, g1b, N, G1_);
  // 5. GCN layer 2: h2 = relu(scatter(bufB @ W2) + self + b2)
  small_matmul<<<(int)(((long)N * G2_ + 255) / 256), 256, 0, stream>>>(bufB, g2W, bufA, N, G1_, G2_);
  zero_f32    <<<(int)(((long)N * G2_ + 255) / 256), 256, 0, stream>>>(h2, (long)N * G2_);
  gcn_scatter <<<(int)((E * (G2_ / 4) + 255) / 256), 256, 0, stream>>>(src, dst, bufA, dinv, h2, E, G2_);
  gcn_finalize<<<(int)(((long)N * G2_ + 255) / 256), 256, 0, stream>>>(h2, bufA, dinv, g2b, N, G2_);
  // 6. gather + FC
  fc_gather<<<(N + 255) / 256, 256, 0, stream>>>(ids, h2, fcW, fcb, (float*)d_out, N);
}
